// AttentionBlock_80032420594248
// MI455X (gfx1250) — matrix operand order repaired and checked by reading, not compiled, not run on hardware
//
#include <hip/hip_runtime.h>

// ---------------------------------------------------------------------------
// AttentionBlock for MI455X (gfx1250, wave32, WMMA bf16 16x16x32, f32 accum)
// B=16, C=512, T=1024, HEADS=8, CH=64, GROUPS=32
// ---------------------------------------------------------------------------

typedef __attribute__((ext_vector_type(16))) __bf16 v16bf;
typedef __attribute__((ext_vector_type(8)))  float  v8f;
typedef int v4i __attribute__((ext_vector_type(4)));   // native vector (not HIP int4)

#define DEV __device__ __forceinline__

#if defined(__has_builtin)
#  if __has_builtin(__builtin_amdgcn_global_load_async_to_lds_b128)
#    define HAS_ASYNC_B128 1
#  endif
#  if __has_builtin(__builtin_amdgcn_s_wait_asynccnt)
#    define HAS_WAIT_ASYNC 1
#  endif
#endif

DEV unsigned short f2bf(float f) {             // fp32->bf16, round-half-up
  union { float f; unsigned u; } x; x.f = f;
  return (unsigned short)((x.u + 0x8000u) >> 16);
}

// pack 2xfp32 -> 2xbf16 in one v_perm_b32 (+2 adds for rounding)
DEV unsigned pk2bf(float lo, float hi) {
  union { float f; unsigned u; } a, b;
  a.f = lo; b.f = hi;
  return __builtin_amdgcn_perm(b.u + 0x8000u, a.u + 0x8000u, 0x07060302u);
}

union U16 { v16bf v; uint4 q[2]; };
union LD8 { uint4 q; unsigned short us[8]; };

// Load a 16-element bf16 fragment as two 16-byte chunks (ds_load_b128)
DEV v16bf ld_frag(const unsigned short* p, int o0, int o1) {
  U16 u;
  u.q[0] = *(const uint4*)(p + o0);
  u.q[1] = *(const uint4*)(p + o1);
  return u.v;
}

DEV v8f bwmma(v16bf a, v16bf b, v8f c) {
  return __builtin_amdgcn_wmma_f32_16x16x32_bf16(false, a, false, b, (short)0, c,
                                                 false, false);
}

// 16-byte global -> LDS copy; async (ASYNCcnt) path when available
DEV void cp_b128_to_lds(unsigned short* lds, const unsigned short* g) {
#if defined(HAS_ASYNC_B128)
  typedef __attribute__((address_space(1))) v4i gv4i;   // global src
  typedef __attribute__((address_space(3))) v4i lv4i;   // LDS dst
  __builtin_amdgcn_global_load_async_to_lds_b128(
      (gv4i*)(unsigned long long)g, (lv4i*)(unsigned long long)lds, 0, 0);
#else
  *(uint4*)lds = *(const uint4*)g;
#endif
}

DEV void async_wait0() {
#if defined(HAS_ASYNC_B128)
#  if defined(HAS_WAIT_ASYNC)
  __builtin_amdgcn_s_wait_asynccnt(0);
#  else
  asm volatile("s_wait_asynccnt 0x0" ::: "memory");
#  endif
#endif
}

// ---------------------------------------------------------------------------
// Kernel 1: GroupNorm (fp32 stats) -> bf16 normalized activations
// grid = B*GROUPS = 512 blocks, 256 threads. Each (b,g) owns 16ch x 1024T.
// ---------------------------------------------------------------------------
__global__ __launch_bounds__(256) void gn_kernel(
    const float* __restrict__ x, const float* __restrict__ scale,
    const float* __restrict__ bias, unsigned short* __restrict__ xn) {
  const int b = blockIdx.x >> 5;
  const int g = blockIdx.x & 31;
  const int tid = threadIdx.x;
  const float4* xp4 = (const float4*)(x + ((size_t)b * 512 + g * 16) * 1024);

  float s = 0.f, s2 = 0.f;
  for (int i = tid; i < 4096; i += 256) {       // 16384 floats as float4
    float4 v = xp4[i];
    s  += v.x + v.y + v.z + v.w;
    s2 += v.x * v.x + v.y * v.y + v.z * v.z + v.w * v.w;
  }
  __shared__ float r0[256], r1[256];
  r0[tid] = s; r1[tid] = s2;
  __syncthreads();
  for (int off = 128; off > 0; off >>= 1) {
    if (tid < off) { r0[tid] += r0[tid + off]; r1[tid] += r1[tid + off]; }
    __syncthreads();
  }
  const float mean = r0[0] * (1.f / 16384.f);
  const float var  = r1[0] * (1.f / 16384.f) - mean * mean;
  const float rinv = __frsqrt_rn(var + 1e-5f);

  uint2* xnp = (uint2*)(xn + ((size_t)b * 512 + g * 16) * 1024);
  for (int i = tid; i < 4096; i += 256) {
    const int cl = i >> 8;                       // channel within group
    const float sc = scale[g * 16 + cl] * rinv;
    const float bi = bias[g * 16 + cl] - mean * sc;
    float4 v = xp4[i];
    uint2 o;
    o.x = pk2bf(v.x * sc + bi, v.y * sc + bi);
    o.y = pk2bf(v.z * sc + bi, v.w * sc + bi);
    xnp[i] = o;
  }
}

// ---------------------------------------------------------------------------
// Shared GEMM tile machinery: 128x128 tile, 8 waves of 32x64, K-step 32.
//   As[o][k]  (fp32 weights converted to bf16)
//   Bs[t][k]  (activations, transposed on the LDS write)
// ---------------------------------------------------------------------------
DEV void stage_w(unsigned short (*As)[40], const float* __restrict__ w,
                 int o0, int kk, int tid) {
  for (int i = tid; i < 1024; i += 256) {        // float4 loads, uint2 stores
    int row = i >> 3, c4 = (i & 7) * 4;
    float4 wv = *(const float4*)&w[(size_t)(o0 + row) * 512 + kk + c4];
    uint2 o;
    o.x = pk2bf(wv.x, wv.y);
    o.y = pk2bf(wv.z, wv.w);
    *(uint2*)&As[row][c4] = o;
  }
}

DEV void stage_act(unsigned short (*Bs)[40], const unsigned short* __restrict__ act,
                   int kk, int tid) {
  for (int i = tid; i < 512; i += 256) {         // uint4 loads, 8-way transpose
    int kr = i >> 4, t8 = (i & 15) * 8;
    LD8 d; d.q = *(const uint4*)&act[(size_t)(kk + kr) * 1024 + t8];
#pragma unroll
    for (int e = 0; e < 8; ++e) Bs[t8 + e][kr] = d.us[e];
  }
}

// ---------------------------------------------------------------------------
// Kernel 2: QKV GEMM [1536x512]x[512x1024] per batch, WMMA bf16.
//   q layout: [b*8+h][t][64]   (A-operand friendly)
//   k layout: [b*8+h][64][t]   (B-operand friendly)
//   v layout: [b*8+h][64][t]
// ---------------------------------------------------------------------------
__global__ __launch_bounds__(256) void qkv_kernel(
    const float* __restrict__ w, const float* __restrict__ bq,
    const unsigned short* __restrict__ xn,
    unsigned short* __restrict__ qo, unsigned short* __restrict__ ko,
    unsigned short* __restrict__ vo) {
  const int t0 = blockIdx.x * 128;
  const int o0 = blockIdx.y * 128;
  const int b  = blockIdx.z;
  const int tid = threadIdx.x, lane = tid & 31, wid = tid >> 5;
  const int half = lane >> 4, lm = lane & 15;
  const int wm = (wid & 3) * 32, wn = (wid >> 2) * 64;

  __shared__ unsigned short As[128][40] __attribute__((aligned(16)));
  __shared__ unsigned short Bs[128][40] __attribute__((aligned(16)));

  v8f acc[2][4] = {};
  const unsigned short* xnb = xn + (size_t)b * 512 * 1024 + t0;

  for (int kk = 0; kk < 512; kk += 32) {
    stage_w(As, w, o0, kk, tid);
    stage_act(Bs, xnb, kk, tid);
    if (kk + 32 < 512) {                          // prefetch next K tile
      __builtin_prefetch(&w[(size_t)(o0 + (tid >> 1)) * 512 + kk + 32 +
                            (tid & 1) * 16], 0, 0);
      __builtin_prefetch(&xnb[(size_t)(kk + 32 + (tid >> 3)) * 1024 +
                              (tid & 7) * 16], 0, 0);
    }
    __syncthreads();
    v16bf af[2], bf[4];
#pragma unroll
    for (int i = 0; i < 2; ++i)
      af[i] = ld_frag(&As[wm + i * 16 + lm][0], half * 8, 16 + half * 8);
#pragma unroll
    for (int j = 0; j < 4; ++j)
      bf[j] = ld_frag(&Bs[wn + j * 16 + lm][0], half * 8, 16 + half * 8);
#pragma unroll
    for (int i = 0; i < 2; ++i)
#pragma unroll
      for (int j = 0; j < 4; ++j) acc[i][j] = bwmma(af[i], bf[j], acc[i][j]);
    __syncthreads();
  }

  const float QKS = 0.35355339059327373f;        // 64^-0.25
#pragma unroll
  for (int i = 0; i < 2; ++i)
#pragma unroll
    for (int j = 0; j < 4; ++j) {
      const int obase = o0 + wm + i * 16 + 8 * half;   // o = obase + r
      const int t = t0 + wn + j * 16 + lm;
      if (o0 < 512) {                             // ---- q: vectorized store
        int h = obase >> 6, c = obase & 63;       // 8 consecutive channels
        float vr[8];
#pragma unroll
        for (int r = 0; r < 8; ++r) vr[r] = (acc[i][j][r] + bq[obase + r]) * QKS;
        uint4 pk;
        pk.x = pk2bf(vr[0], vr[1]); pk.y = pk2bf(vr[2], vr[3]);
        pk.z = pk2bf(vr[4], vr[5]); pk.w = pk2bf(vr[6], vr[7]);
        *(uint4*)&qo[(((size_t)b * 8 + h) * 1024 + t) * 64 + c] = pk;
      } else if (o0 < 1024) {                     // ---- k
#pragma unroll
        for (int r = 0; r < 8; ++r) {
          int oo = obase + r - 512, h = oo >> 6, c = oo & 63;
          ko[(((size_t)b * 8 + h) * 64 + c) * 1024 + t] =
              f2bf((acc[i][j][r] + bq[obase + r]) * QKS);
        }
      } else {                                    // ---- v (no scale)
#pragma unroll
        for (int r = 0; r < 8; ++r) {
          int oo = obase + r - 1024, h = oo >> 6, c = oo & 63;
          vo[(((size_t)b * 8 + h) * 64 + c) * 1024 + t] =
              f2bf(acc[i][j][r] + bq[obase + r]);
        }
      }
    }
}

// ---------------------------------------------------------------------------
// Kernel 3: Flash attention per (b, h, 128-query tile). 8 waves x 16 rows.
// Q fragments loaded once from global (native A layout). Per 64-wide s-block:
// stage K (transpose) and V (async b128 copy) -> LDS, S = Q@K^T (WMMA),
// online softmax in C-fragment layout, P -> LDS, O += P@V (WMMA).
// ---------------------------------------------------------------------------
__global__ __launch_bounds__(256) void attn_kernel(
    const unsigned short* __restrict__ q, const unsigned short* __restrict__ k,
    const unsigned short* __restrict__ v, unsigned short* __restrict__ ao) {
  const int t0 = blockIdx.x * 128;
  const int bh = blockIdx.y;
  const int b = bh >> 3, h = bh & 7;
  const int tid = threadIdx.x, lane = tid & 31, wid = tid >> 5;
  const int half = lane >> 4, lm = lane & 15;

  __shared__ unsigned short Ks[64][72]    __attribute__((aligned(16))); // [s][c]
  __shared__ unsigned short Vs[64][72]    __attribute__((aligned(16))); // [c][s]
  __shared__ unsigned short Ps[8][16][72] __attribute__((aligned(16))); // [w][m][s]

  const unsigned short* qp = q + ((size_t)bh * 1024 + t0 + wid * 16 + lm) * 64;
  v16bf qf[2];
  qf[0] = ld_frag(qp, half * 8, 16 + half * 8);
  qf[1] = ld_frag(qp, 32 + half * 8, 48 + half * 8);

  v8f oacc[4] = {};
  float mrow[8], lrow[8];
#pragma unroll
  for (int r = 0; r < 8; ++r) { mrow[r] = -1e30f; lrow[r] = 0.f; }

  const unsigned short* kp = k + (size_t)bh * 64 * 1024;
  const unsigned short* vp = v + (size_t)bh * 64 * 1024;

  for (int sb = 0; sb < 16; ++sb) {
    const int s0 = sb * 64;
    for (int i = tid; i < 512; i += 256) {        // V: contiguous 16B copies
      int c = i >> 3, s8 = (i & 7) * 8;
      cp_b128_to_lds(&Vs[c][s8], &vp[(size_t)c * 1024 + s0 + s8]);
    }
    for (int i = tid; i < 512; i += 256) {        // K: uint4 load + transpose
      int c = i >> 3, s8 = (i & 7) * 8;
      LD8 d; d.q = *(const uint4*)&kp[(size_t)c * 1024 + s0 + s8];
#pragma unroll
      for (int e = 0; e < 8; ++e) Ks[s8 + e][c] = d.us[e];
    }
    async_wait0();
    __syncthreads();

    v8f sacc[4] = {};
#pragma unroll
    for (int ch = 0; ch < 2; ++ch)
#pragma unroll
      for (int j = 0; j < 4; ++j) {
        v16bf kf = ld_frag(&Ks[j * 16 + lm][0], ch * 32 + half * 8,
                           ch * 32 + 16 + half * 8);
        sacc[j] = bwmma(qf[ch], kf, sacc[j]);
      }

    // online softmax; row m = r + 8*half lives across the 16 lanes of a half
#pragma unroll
    for (int r = 0; r < 8; ++r) {
      float rm = fmaxf(fmaxf(sacc[0][r], sacc[1][r]),
                       fmaxf(sacc[2][r], sacc[3][r]));
#pragma unroll
      for (int m = 1; m < 16; m <<= 1) rm = fmaxf(rm, __shfl_xor(rm, m, 32));
      float mnew = fmaxf(mrow[r], rm);
      float corr = __expf(mrow[r] - mnew);
      float rs = 0.f;
#pragma unroll
      for (int j = 0; j < 4; ++j) {
        float p = __expf(sacc[j][r] - mnew);
        rs += p;
        Ps[wid][r + 8 * half][j * 16 + lm] = f2bf(p);
      }
#pragma unroll
      for (int m = 1; m < 16; m <<= 1) rs += __shfl_xor(rs, m, 32);
      lrow[r] = lrow[r] * corr + rs;
      mrow[r] = mnew;
#pragma unroll
      for (int j = 0; j < 4; ++j) oacc[j][r] *= corr;
    }
    __syncthreads();

#pragma unroll
    for (int ch = 0; ch < 2; ++ch) {
      v16bf pf = ld_frag(&Ps[wid][lm][0], ch * 32 + half * 8,
                         ch * 32 + 16 + half * 8);
#pragma unroll
      for (int j = 0; j < 4; ++j) {
        v16bf vf = ld_frag(&Vs[j * 16 + lm][0], ch * 32 + half * 8,
                           ch * 32 + 16 + half * 8);
        oacc[j] = bwmma(pf, vf, oacc[j]);
      }
    }
    __syncthreads();
  }

#pragma unroll
  for (int j = 0; j < 4; ++j) {                   // 8 consecutive t per reg
    int c = h * 64 + j * 16 + lm;
    int tbase = t0 + wid * 16 + 8 * half;
    float inv = 1.f;                              // lrow identical across half
    float vr[8];
#pragma unroll
    for (int r = 0; r < 8; ++r) vr[r] = oacc[j][r] / lrow[r];
    (void)inv;
    uint4 pk;
    pk.x = pk2bf(vr[0], vr[1]); pk.y = pk2bf(vr[2], vr[3]);
    pk.z = pk2bf(vr[4], vr[5]); pk.w = pk2bf(vr[6], vr[7]);
    *(uint4*)&ao[((size_t)b * 512 + c) * 1024 + tbase] = pk;
  }
}

// ---------------------------------------------------------------------------
// Kernel 4: proj GEMM [512x512]x[512x1024] + bias + residual, fp32 out.
// ---------------------------------------------------------------------------
__global__ __launch_bounds__(256) void proj_kernel(
    const float* __restrict__ w, const float* __restrict__ bp,
    const unsigned short* __restrict__ a, const float* __restrict__ x,
    float* __restrict__ out) {
  const int t0 = blockIdx.x * 128;
  const int o0 = blockIdx.y * 128;
  const int b  = blockIdx.z;
  const int tid = threadIdx.x, lane = tid & 31, wid = tid >> 5;
  const int half = lane >> 4, lm = lane & 15;
  const int wm = (wid & 3) * 32, wn = (wid >> 2) * 64;

  __shared__ unsigned short As[128][40] __attribute__((aligned(16)));
  __shared__ unsigned short Bs[128][40] __attribute__((aligned(16)));

  v8f acc[2][4] = {};
  const unsigned short* ab = a + (size_t)b * 512 * 1024 + t0;

  for (int kk = 0; kk < 512; kk += 32) {
    stage_w(As, w, o0, kk, tid);
    stage_act(Bs, ab, kk, tid);
    if (kk + 32 < 512) {
      __builtin_prefetch(&w[(size_t)(o0 + (tid >> 1)) * 512 + kk + 32 +
                            (tid & 1) * 16], 0, 0);
      __builtin_prefetch(&ab[(size_t)(kk + 32 + (tid >> 3)) * 1024 +
                             (tid & 7) * 16], 0, 0);
    }
    __syncthreads();
    v16bf af[2], bf[4];
#pragma unroll
    for (int i = 0; i < 2; ++i)
      af[i] = ld_frag(&As[wm + i * 16 + lm][0], half * 8, 16 + half * 8);
#pragma unroll
    for (int j = 0; j < 4; ++j)
      bf[j] = ld_frag(&Bs[wn + j * 16 + lm][0], half * 8, 16 + half * 8);
#pragma unroll
    for (int i = 0; i < 2; ++i)
#pragma unroll
      for (int j = 0; j < 4; ++j) acc[i][j] = bwmma(af[i], bf[j], acc[i][j]);
    __syncthreads();
  }

#pragma unroll
  for (int i = 0; i < 2; ++i)
#pragma unroll
    for (int j = 0; j < 4; ++j)
#pragma unroll
      for (int r = 0; r < 8; ++r) {
        int o = o0 + wm + i * 16 + r + 8 * half;
        int t = t0 + wn + j * 16 + lm;
        size_t idx = ((size_t)b * 512 + o) * 1024 + t;
        out[idx] = x[idx] + acc[i][j][r] + bp[o];
      }
}

// ---------------------------------------------------------------------------
extern "C" void kernel_launch(void* const* d_in, const int* in_sizes, int n_in,
                              void* d_out, int out_size, void* d_ws,
                              size_t ws_size, hipStream_t stream) {
  (void)in_sizes; (void)n_in; (void)out_size; (void)ws_size;
  const float* x   = (const float*)d_in[0];
  const float* gsc = (const float*)d_in[1];
  const float* gbi = (const float*)d_in[2];
  const float* wq  = (const float*)d_in[3];
  const float* bq  = (const float*)d_in[4];
  const float* wp  = (const float*)d_in[5];
  const float* bp  = (const float*)d_in[6];
  float* out = (float*)d_out;

  const size_t N = (size_t)16 * 512 * 1024;      // 8M elements
  unsigned short* xn = (unsigned short*)d_ws;    // bf16 normalized x
  unsigned short* qw = xn + N;                   // q  [bh][t][c]
  unsigned short* kw = qw + N;                   // k  [bh][c][t]
  unsigned short* vw = kw + N;                   // v  [bh][c][t]
  unsigned short* aw = xn;                       // reuse xn for attn output

  gn_kernel  <<<dim3(512),        dim3(256), 0, stream>>>(x, gsc, gbi, xn);
  qkv_kernel <<<dim3(8, 12, 16),  dim3(256), 0, stream>>>(wq, bq, xn, qw, kw, vw);
  attn_kernel<<<dim3(8, 128),     dim3(256), 0, stream>>>(qw, kw, vw, aw);
  proj_kernel<<<dim3(8, 4, 16),   dim3(256), 0, stream>>>(wp, bp, aw, x, out);
}
